// HyperQMDPLayer_11982958755879
// MI455X (gfx1250) — matrix operand order repaired and checked by reading, not compiled, not run on hardware
//
#include <hip/hip_runtime.h>
#include <hip/hip_bf16.h>
#include <stdint.h>

#define S_DIM 64
#define A_DIM 16
#define R_DIM 16
#define H_DIM 30
#define HYP 64
#define T_STEPS 1000
#define B_BATCH 256
#define EPSV 1e-6f
#define LOG_LOW -13.815510558f
#define LOG_HIGH 13.815510558f

typedef __attribute__((ext_vector_type(16))) __bf16 v16bf;
typedef __attribute__((ext_vector_type(8)))  __bf16 v8bf;
typedef __attribute__((ext_vector_type(8)))  float  v8f;

// ---------------------------------------------------------------------------
// Kernel A: one workgroup per batch n. Build transition[n] (256 KB) in LDS,
// run value iteration (29 steps) writing value levels as bf16 to ws, then run
// the 1000-step belief recursion entirely out of LDS, writing alpha_b to out.
// ---------------------------------------------------------------------------
extern "C" __global__ void __launch_bounds__(256, 1)
qmdp_beliefs(const float* __restrict__ logp_o, const float* __restrict__ logp_u,
             const float* __restrict__ reward, const float* __restrict__ z,
             const float* __restrict__ b0, const float* __restrict__ gamma_w,
             const float* __restrict__ gamma_b, const float* __restrict__ u,
             const float* __restrict__ v, const float* __restrict__ w,
             float* __restrict__ alpha_b, __hip_bfloat16* __restrict__ value_ws)
{
    extern __shared__ float sm[];
    float* Tl  = sm;                          // [A][S][S] = 65536 f32 (256 KB)
    float* q   = Tl + A_DIM * S_DIM * S_DIM;  // [A][S]
    float* vn  = q  + A_DIM * S_DIM;          // [S]
    float* bv  = vn + S_DIM;                  // [S]
    float* av  = bv + S_DIM;                  // [A]  raw logp_u staging
    float* av2 = av + A_DIM;                  // [A]  normalized action probs
    float* sp  = av2 + A_DIM;                 // [4][S]
    float* xr  = sp + 4 * S_DIM;              // [S]
    float* ul  = xr + S_DIM;                  // [R][S]
    float* vl  = ul + R_DIM * S_DIM;          // [R][S]
    float* wl  = vl + R_DIM * S_DIM;          // [R][A]

    const int n    = blockIdx.x;
    const int tid  = threadIdx.x;
    const int lane = tid & 31;

    // Stage rank factors into LDS.
    for (int e = tid; e < R_DIM * S_DIM; e += 256) { ul[e] = u[e]; vl[e] = v[e]; }
    for (int e = tid; e < R_DIM * A_DIM; e += 256) wl[e] = w[e];
    __syncthreads();

    // Phase 0a: w_full[k][i][j] = sum_r w[r,k] u[r,i] v[r,j] directly into Tl.
    for (int e = tid; e < A_DIM * S_DIM * S_DIM; e += 256) {
        int j = e & 63, i = (e >> 6) & 63, k = e >> 12;
        float acc = 0.f;
        #pragma unroll
        for (int r = 0; r < R_DIM; ++r)
            acc += wl[r * A_DIM + k] * ul[r * S_DIM + i] * vl[r * S_DIM + j];
        Tl[e] = acc;
    }
    __syncthreads();

    // Phase 0b: per-row gamma scale + softmax over j, in place.
    for (int row = tid; row < A_DIM * S_DIM; row += 256) {
        const float* gw = gamma_w + (size_t)row * HYP;
        const float* zn = z + (size_t)n * HYP;
        float g = gamma_b[row];
        #pragma unroll 4
        for (int hh = 0; hh < HYP; ++hh) g += zn[hh] * gw[hh];
        g = __expf(fminf(fmaxf(g, LOG_LOW), LOG_HIGH));
        float* rp = Tl + row * S_DIM;
        float m = -3.4e38f;
        for (int jj = 0; jj < S_DIM; ++jj) { int j = (jj + lane) & 63; m = fmaxf(m, g * rp[j]); }
        float ssum = 0.f;
        for (int jj = 0; jj < S_DIM; ++jj) {
            int j = (jj + lane) & 63;
            float e = __expf(g * rp[j] - m);
            rp[j] = e; ssum += e;
        }
        float inv = 1.f / ssum;
        for (int jj = 0; jj < S_DIM; ++jj) { int j = (jj + lane) & 63; rp[j] *= inv; }
    }
    __syncthreads();

    // Phase 1: value iteration, levels stored to ws as bf16 [n][h][k*S+i].
    __hip_bfloat16* vws = value_ws + (size_t)n * H_DIM * A_DIM * S_DIM;
    for (int e = tid; e < A_DIM * S_DIM; e += 256) {
        float rv = reward[(size_t)n * A_DIM * S_DIM + e];
        q[e] = rv;
        vws[e] = __float2bfloat16(rv);
    }
    __syncthreads();
    for (int h = 1; h < H_DIM; ++h) {
        if (tid < S_DIM) {
            float m = -3.4e38f;
            #pragma unroll
            for (int k = 0; k < A_DIM; ++k) m = fmaxf(m, q[k * S_DIM + tid]);
            float ssum = 0.f;
            #pragma unroll
            for (int k = 0; k < A_DIM; ++k) ssum += __expf(q[k * S_DIM + tid] - m);
            vn[tid] = m + __logf(ssum);
        }
        __syncthreads();
        float nq[4];
        #pragma unroll
        for (int rr = 0; rr < 4; ++rr) {
            int row = tid * 4 + rr;            // row = k*64 + i
            const float* tp = Tl + row * S_DIM;
            float acc = reward[(size_t)n * A_DIM * S_DIM + row];
            for (int jj = 0; jj < S_DIM; ++jj) { int j = (jj + lane) & 63; acc += tp[j] * vn[j]; }
            nq[rr] = acc;
        }
        __syncthreads();
        #pragma unroll
        for (int rr = 0; rr < 4; ++rr) {
            int row = tid * 4 + rr;
            q[row] = nq[rr];
            vws[h * A_DIM * S_DIM + row] = __float2bfloat16(nq[rr]);
        }
        __syncthreads();
    }

    // Phase 2: belief recursion, 1000 sequential steps out of LDS.
    if (tid < S_DIM) bv[tid] = b0[(size_t)n * S_DIM + tid];
    __syncthreads();

    const int jcol = tid & 63;   // output column (LDS bank)
    const int kg   = tid >> 6;   // action group (4 actions each)

    for (int t = 0; t < T_STEPS; ++t) {
        // Hide next step's global latency (global_prefetch_b8).
        if (t + 1 < T_STEPS) {
            if (tid == 0)  __builtin_prefetch(&logp_u[((size_t)(t + 1) * B_BATCH + n) * A_DIM], 0, 1);
            if (tid == 32) __builtin_prefetch(&logp_o[((size_t)(t + 1) * B_BATCH + n) * S_DIM], 0, 1);
        }
        if (tid < A_DIM) av[tid] = logp_u[((size_t)t * B_BATCH + n) * A_DIM + tid];
        __syncthreads();
        if (tid < A_DIM) {           // wave 0 lanes 0-15: lockstep reads, disjoint writes
            float m = -3.4e38f;
            #pragma unroll
            for (int k = 0; k < A_DIM; ++k) m = fmaxf(m, av[k]);
            float ssum = 0.f;
            #pragma unroll
            for (int k = 0; k < A_DIM; ++k) ssum += __expf(av[k] - m);
            av2[tid] = __expf(av[tid] - m) / ssum;
        }
        __syncthreads();

        // s[j] partial over 4 actions; tp[i*S] hits bank jcol only -> conflict free
        float acc = 0.f;
        #pragma unroll
        for (int kk = 0; kk < 4; ++kk) {
            int k = kg * 4 + kk;
            const float* tp = Tl + (size_t)k * S_DIM * S_DIM + jcol;
            float inner = 0.f;
            #pragma unroll 8
            for (int i = 0; i < S_DIM; ++i) inner += bv[i] * tp[i * S_DIM];
            acc += av2[k] * inner;
        }
        sp[kg * S_DIM + jcol] = acc;
        __syncthreads();
        if (tid < S_DIM) {
            float s = sp[tid] + sp[S_DIM + tid] + sp[2 * S_DIM + tid] + sp[3 * S_DIM + tid];
            xr[tid] = __logf(s + EPSV) + logp_o[((size_t)t * B_BATCH + n) * S_DIM + tid];
        }
        __syncthreads();
        float bnew = 0.f;
        if (tid < S_DIM) {
            float m = -3.4e38f;
            for (int j = 0; j < S_DIM; ++j) m = fmaxf(m, xr[j]);
            float ssum = 0.f;
            for (int j = 0; j < S_DIM; ++j) ssum += __expf(xr[j] - m);
            bnew = __expf(xr[tid] - m) / ssum;
        }
        __syncthreads();
        if (tid < S_DIM) {
            bv[tid] = bnew;
            alpha_b[((size_t)t * B_BATCH + n) * S_DIM + tid] = bnew;
        }
        __syncthreads();
    }
}

// ---------------------------------------------------------------------------
// Kernel B: planning GEMM with bf16 WMMA, transposed so softmax-over-k is
// (mostly) in-lane:  D = value_h (M=16 k x K=64 i)  x  alpha_b^T (K=64 i x N=16 t)
// D fragment: M=k lives in the 8 accumulator VGPRs (+ wave-half), N=t = lane.
// Softmax over k = 7 in-lane max/add + ONE xor-16 lane exchange per reduction.
// B fragments (alpha_b) are loop-invariant: loaded from global once per tile.
// ---------------------------------------------------------------------------
extern "C" __global__ void __launch_bounds__(256, 1)
qmdp_plan(const float* __restrict__ alpha_b, const __hip_bfloat16* __restrict__ value_ws,
          const float* __restrict__ z, const float* __restrict__ tau_w,
          const float* __restrict__ tau_b, float* __restrict__ alpha_pi)
{
    __shared__ __attribute__((aligned(32))) __bf16 Vl[H_DIM * A_DIM * S_DIM];
    __shared__ float tau_l[H_DIM];

    const int n    = blockIdx.x;
    const int tid  = threadIdx.x;
    const int wave = tid >> 5;
    const int lane = tid & 31;

    // Stage value[n] (bf16) into LDS as 32-bit copies.
    {
        const uint32_t* src = (const uint32_t*)(value_ws + (size_t)n * H_DIM * A_DIM * S_DIM);
        uint32_t* dst = (uint32_t*)Vl;
        for (int e = tid; e < H_DIM * A_DIM * S_DIM / 2; e += 256) dst[e] = src[e];
    }
    // Truncated-Poisson horizon weights tau[n][0..29].
    if (tid == 0) {
        float r = tau_b[0];
        #pragma unroll 4
        for (int hh = 0; hh < HYP; ++hh) r += z[(size_t)n * HYP + hh] * tau_w[hh];
        r = __expf(fminf(fmaxf(r, LOG_LOW), LOG_HIGH));
        float lr = __logf(r);
        float lfact = 0.f, m = -3.4e38f;
        float lp[H_DIM];
        for (int k = 1; k <= H_DIM; ++k) {
            lfact += __logf((float)k);             // log(k!)
            lp[k - 1] = (float)k * lr - r - lfact;
            m = fmaxf(m, lp[k - 1]);
        }
        float ssum = 0.f;
        for (int k = 0; k < H_DIM; ++k) ssum += __expf(lp[k] - m);
        float inv = 1.f / ssum;
        for (int k = 0; k < H_DIM; ++k) tau_l[k] = __expf(lp[k] - m) * inv;
    }
    __syncthreads();

    // ---- B fragments: alpha_b tile, K=64 (i) x N=16 (t), loaded ONCE. ----
    // ISA 7.12.2 16-bit B layout: lane = N column; lanes 0-15 hold K {0-7,16-23},
    // lanes 16-31 hold K {8-15,24-31} (frag1: +32).
    const int t0 = (blockIdx.y * 8 + wave) * 16;
    const int nn = lane & 15;                    // N column == timestep in tile
    int trow = t0 + nn;
    trow = trow < T_STEPS ? trow : T_STEPS - 1;
    const float4* r4 = (const float4*)(alpha_b + ((size_t)trow * B_BATCH + n) * S_DIM);
    const int kq = (lane < 16) ? 0 : 2;          // kofs/4

    v16bf b0, b1;
#define PACK4(A, base, f) { A[(base)] = (__bf16)(f).x; A[(base)+1] = (__bf16)(f).y; \
                            A[(base)+2] = (__bf16)(f).z; A[(base)+3] = (__bf16)(f).w; }
    PACK4(b0, 0,  r4[kq]);     PACK4(b0, 4,  r4[kq + 1]);
    PACK4(b0, 8,  r4[kq + 4]); PACK4(b0, 12, r4[kq + 5]);
    PACK4(b1, 0,  r4[kq + 8]); PACK4(b1, 4,  r4[kq + 9]);
    PACK4(b1, 8,  r4[kq + 12]);PACK4(b1, 12, r4[kq + 13]);
#undef PACK4

    // ---- A fragments per chunk: value_h, M=16 (k) x K=64 (i). ----
    // ISA 7.12.2 16-bit A layout: lanes 0-15: M=lane, K {0-7,16-23};
    // lanes 16-31: M=lane-16, K {8-15,24-31}.  (frag1: +32)
    const int mrow  = lane & 15;                 // M row == action k
    const int kbase = (lane < 16) ? 0 : 8;

    float accpi[8];
    #pragma unroll
    for (int r = 0; r < 8; ++r) accpi[r] = 0.f;

    for (int c = 0; c < H_DIM; ++c) {            // one chunk == one horizon h
        const __bf16* vp = Vl + (c * A_DIM + mrow) * S_DIM + kbase;
        v8bf a0lo = *(const v8bf*)(vp);          // i = kbase .. +7
        v8bf a0hi = *(const v8bf*)(vp + 16);     // i = kbase+16 .. +23
        v8bf a1lo = *(const v8bf*)(vp + 32);
        v8bf a1hi = *(const v8bf*)(vp + 48);
        v16bf a0 = __builtin_shufflevector(a0lo, a0hi, 0,1,2,3,4,5,6,7,8,9,10,11,12,13,14,15);
        v16bf a1 = __builtin_shufflevector(a1lo, a1hi, 0,1,2,3,4,5,6,7,8,9,10,11,12,13,14,15);

        v8f d = {0.f, 0.f, 0.f, 0.f, 0.f, 0.f, 0.f, 0.f};
        d = __builtin_amdgcn_wmma_f32_16x16x32_bf16(false, a0, false, b0, (short)0, d, false, false);
        d = __builtin_amdgcn_wmma_f32_16x16x32_bf16(false, a1, false, b1, (short)0, d, false, false);

        // D: lane holds k = r + 8*(lane>=16) for fixed t = lane&15.
        // Softmax over k: in-lane over 8 regs + one cross-half exchange.
        float mx = d[0];
        #pragma unroll
        for (int r = 1; r < 8; ++r) mx = fmaxf(mx, d[r]);
        mx = fmaxf(mx, __shfl_xor(mx, 16, 32));
        float ex[8], se = 0.f;
        #pragma unroll
        for (int r = 0; r < 8; ++r) { ex[r] = __expf(d[r] - mx); se += ex[r]; }
        se += __shfl_xor(se, 16, 32);
        float sc = tau_l[c] / se;
        #pragma unroll
        for (int r = 0; r < 8; ++r) accpi[r] += ex[r] * sc;
    }

    // Store alpha_pi[t, n, k]: lane owns t = t0+nn, k = (lane>=16?8:0)+r
    // -> 8 consecutive floats: two float4 stores.
    const int t = t0 + nn;
    if (t < T_STEPS) {
        float* p = alpha_pi + ((size_t)t * B_BATCH + n) * A_DIM + ((lane >> 4) << 3);
        float4 lo = {accpi[0], accpi[1], accpi[2], accpi[3]};
        float4 hi = {accpi[4], accpi[5], accpi[6], accpi[7]};
        *(float4*)p = lo;
        *(float4*)(p + 4) = hi;
    }
}

// ---------------------------------------------------------------------------
extern "C" void kernel_launch(void* const* d_in, const int* in_sizes, int n_in,
                              void* d_out, int out_size, void* d_ws, size_t ws_size,
                              hipStream_t stream) {
    const float* logp_o  = (const float*)d_in[0];
    const float* logp_u  = (const float*)d_in[1];
    const float* reward  = (const float*)d_in[2];
    const float* z       = (const float*)d_in[3];
    const float* b0      = (const float*)d_in[4];
    const float* tau_w   = (const float*)d_in[5];
    const float* tau_b   = (const float*)d_in[6];
    const float* gamma_w = (const float*)d_in[7];
    const float* gamma_b = (const float*)d_in[8];
    const float* u       = (const float*)d_in[9];
    const float* v       = (const float*)d_in[10];
    const float* w       = (const float*)d_in[11];

    float* out_alpha_b  = (float*)d_out;
    float* out_alpha_pi = out_alpha_b + (size_t)T_STEPS * B_BATCH * S_DIM;
    __hip_bfloat16* vws = (__hip_bfloat16*)d_ws;

    const size_t ldsA = (size_t)(A_DIM * S_DIM * S_DIM + A_DIM * S_DIM + S_DIM + S_DIM +
                                 2 * A_DIM + 4 * S_DIM + S_DIM + 2 * R_DIM * S_DIM +
                                 R_DIM * A_DIM) * sizeof(float);   // ~271 KB < 320 KB WGP LDS
    hipFuncSetAttribute((const void*)qmdp_beliefs,
                        hipFuncAttributeMaxDynamicSharedMemorySize, (int)ldsA);

    qmdp_beliefs<<<dim3(B_BATCH), dim3(256), ldsA, stream>>>(
        logp_o, logp_u, reward, z, b0, gamma_w, gamma_b, u, v, w, out_alpha_b, vws);

    qmdp_plan<<<dim3(B_BATCH, 8), dim3(256), 0, stream>>>(
        out_alpha_b, vws, z, tau_w, tau_b, out_alpha_pi);
}
